// StyleNeuSLXY_21242908246589
// MI455X (gfx1250) — matrix operand order repaired and checked by reading, not compiled, not run on hardware
//
#include <hip/hip_runtime.h>
#include <math.h>

#define RES    512
#define NFEAT  32
#define TPW    4            // tiles (of 16 points) per wave
#define WAVES  8            // waves per block (256 threads)

typedef __attribute__((ext_vector_type(16))) _Float16 v16h;
typedef __attribute__((ext_vector_type(8)))  float    v8f;
typedef __attribute__((ext_vector_type(2)))  _Float16 h2;

__device__ __forceinline__ float4 ld4(const float* __restrict__ p, int c) {
    return reinterpret_cast<const float4*>(p)[c];
}

// weight-fragment table indices in LDS
#define FR_W1LO  0
#define FR_W1HI  1
#define FR_W1TLO 2
#define FR_W1THI 3
#define FR_W2    4
#define FR_W2C0  5

__global__ __launch_bounds__(256, 4)
void triplane_fused_kernel(const float* __restrict__ x,
                           const float* __restrict__ planes,
                           const float* __restrict__ W1,
                           const float* __restrict__ b1,
                           const float* __restrict__ W2,
                           const float* __restrict__ b2,
                           float* __restrict__ out,
                           long N)
{
    const int lane  = threadIdx.x & 31;
    const int wid   = threadIdx.x >> 5;        // wave in block: 0..7
    const int m     = lane & 15;               // point row within tile / column id
    const int halfk = (lane >> 4) & 1;         // which 16-feature half this lane owns

    // per-wave LDS tiles (private per wave; barriers order same-wave store->load phases)
    __shared__ _Float16 sH[WAVES][16 * 34];    // f16 tile (h, then y1 pre-activation)
    __shared__ float    sG[WAVES][16 * 33];    // f32 tile (dsdf/dh)
    // weight fragments, shared by all waves: [frag][lane][16 f16] = 6 KB
    __shared__ __align__(16) _Float16 sW[6][32][16];

    const float s = 0.17677669529663687f;      // 1/sqrt(NFEAT)

    const float b1a = b1[m];
    const float b1b = b1[m + 16];
    const float b2n = (m < 4) ? b2[m] : 0.0f;

    // ---- wave 0 builds the per-lane weight fragments once (B layout: K=(e>>3)*16+halfk*8+(e&7), N=m)
    if (wid == 0) {
        v16h f0, f1, f2, f3, f4, f5;
#pragma unroll
        for (int e = 0; e < 16; ++e) {
            const int K = (e >> 3) * 16 + halfk * 8 + (e & 7);
            f0[e] = (_Float16)(W1[K * 32 + m] * s);            // y1 cols 0..15
            f1[e] = (_Float16)(W1[K * 32 + 16 + m] * s);       // y1 cols 16..31
            f2[e] = (_Float16)(W1[m * 32 + K] * s);            // W1^T cols 0..15
            f3[e] = (_Float16)(W1[(16 + m) * 32 + K] * s);     // W1^T cols 16..31
            f4[e] = (m < 4) ? (_Float16)(W2[K * 4 + m] * s) : (_Float16)0.0f;
            // A-layout element e -> k index (D matrix is scaled by W2 column 0)
            const int i  = e >> 1;
            const int ke = (i >> 2) * 16 + halfk * 8 + (i & 3) * 2 + (e & 1);
            f5[e] = (_Float16)(W2[ke * 4 + 0] * s);
        }
        *reinterpret_cast<v16h*>(&sW[FR_W1LO ][lane][0]) = f0;
        *reinterpret_cast<v16h*>(&sW[FR_W1HI ][lane][0]) = f1;
        *reinterpret_cast<v16h*>(&sW[FR_W1TLO][lane][0]) = f2;
        *reinterpret_cast<v16h*>(&sW[FR_W1THI][lane][0]) = f3;
        *reinterpret_cast<v16h*>(&sW[FR_W2   ][lane][0]) = f4;
        *reinterpret_cast<v16h*>(&sW[FR_W2C0 ][lane][0]) = f5;
    }
    __syncthreads();

    float* outSdf = out;
    float* outH   = out + N;
    float* outNab = out + 33 * N;
    float* outRgb = out + 36 * N;

    const long waveGlobal = (long)blockIdx.x * WAVES + wid;
    const int axA[3] = {0, 0, 1};
    const int axB[3] = {1, 2, 2};

    for (int t = 0; t < TPW; ++t) {
        const long base   = (waveGlobal * TPW + t) * 16;
        const bool active = (base < N);
        long p = base + m; if (p >= N) p = N - 1;

        int   P0[3], P1[3];
        float F[3];
        if (active) {
#pragma unroll
            for (int d = 0; d < 3; ++d) {
                float xt  = (x[p * 3 + d] + 1.0f) * 0.5f;
                float pos = xt * (float)(RES - 1);
                float pf  = floorf(pos);
                int i0 = (int)pf;
                i0 = i0 < 0 ? 0 : (i0 > RES - 1 ? RES - 1 : i0);
                int i1 = i0 + 1; if (i1 > RES - 1) i1 = RES - 1;
                P0[d] = i0; P1[d] = i1; F[d] = pos - pf;
            }
        }

        // ---------------- phase 1: gather + encode (h), stage f16 tile --------------
        if (active) {
            float hA[16];
#pragma unroll
            for (int f = 0; f < 16; ++f) hA[f] = 0.0f;
#pragma unroll
            for (int pl = 0; pl < 3; ++pl) {
                const int a = axA[pl], b = axB[pl];
                const float fa = F[a], fb = F[b];
                const float* pb  = planes + pl * RES * RES * NFEAT + halfk * 16;
                const float* r00 = pb + (P0[a] * RES + P0[b]) * NFEAT;
                const float* r01 = pb + (P0[a] * RES + P1[b]) * NFEAT;
                const float* r10 = pb + (P1[a] * RES + P0[b]) * NFEAT;
                const float* r11 = pb + (P1[a] * RES + P1[b]) * NFEAT;
                const float w00 = (1.f - fa) * (1.f - fb), w01 = (1.f - fa) * fb;
                const float w10 = fa * (1.f - fb),         w11 = fa * fb;
#pragma unroll
                for (int c = 0; c < 4; ++c) {          // 4-feature chunks: 16 live floats
                    float4 a00 = ld4(r00, c), a01 = ld4(r01, c);
                    float4 a10 = ld4(r10, c), a11 = ld4(r11, c);
                    hA[4*c+0] += a00.x*w00 + a01.x*w01 + a10.x*w10 + a11.x*w11;
                    hA[4*c+1] += a00.y*w00 + a01.y*w01 + a10.y*w10 + a11.y*w11;
                    hA[4*c+2] += a00.z*w00 + a01.z*w01 + a10.z*w10 + a11.z*w11;
                    hA[4*c+3] += a00.w*w00 + a01.w*w01 + a10.w*w10 + a11.w*w11;
                }
            }
            if (base + m < N) {   // h output [N,32]
                float4* ho = reinterpret_cast<float4*>(outH + (base + m) * 32 + halfk * 16);
                ho[0] = make_float4(hA[0],  hA[1],  hA[2],  hA[3]);
                ho[1] = make_float4(hA[4],  hA[5],  hA[6],  hA[7]);
                ho[2] = make_float4(hA[8],  hA[9],  hA[10], hA[11]);
                ho[3] = make_float4(hA[12], hA[13], hA[14], hA[15]);
            }
#pragma unroll
            for (int f = 0; f < 16; f += 2) {
                h2 pr; pr[0] = (_Float16)hA[f]; pr[1] = (_Float16)hA[f + 1];
                *reinterpret_cast<h2*>(&sH[wid][m * 34 + halfk * 16 + f]) = pr;
            }
        }
        __syncthreads();

        // ---------------- phase 2: y1 = h @ (W1*s)  (two WMMAs) --------------------
        v8f c0 = {}, c1 = {};
        if (active) {
            v16h A;
#pragma unroll
            for (int i = 0; i < 8; ++i) {
                const int k0 = (i >> 2) * 16 + halfk * 8 + (i & 3) * 2;
                h2 pr = *reinterpret_cast<const h2*>(&sH[wid][m * 34 + k0]);
                A[2 * i] = pr[0]; A[2 * i + 1] = pr[1];
            }
            v16h bw1lo = *reinterpret_cast<const v16h*>(&sW[FR_W1LO][lane][0]);
            v16h bw1hi = *reinterpret_cast<const v16h*>(&sW[FR_W1HI][lane][0]);
            c0 = __builtin_amdgcn_wmma_f32_16x16x32_f16(false, A, false, bw1lo, (short)0, c0, false, false);
            c1 = __builtin_amdgcn_wmma_f32_16x16x32_f16(false, A, false, bw1hi, (short)0, c1, false, false);
        }
        __syncthreads();

        // stage y1 pre-activation (f16) back to the tile
        if (active) {
#pragma unroll
            for (int r = 0; r < 8; ++r) {
                const int M = r + halfk * 8;
                sH[wid][M * 34 + m]      = (_Float16)(c0[r] + b1a);
                sH[wid][M * 34 + m + 16] = (_Float16)(c1[r] + b1b);
            }
        }
        __syncthreads();

        // -------- phase 3: out = lrelu(y1) @ (W2*s); G = D @ (W1*s)^T (3 WMMAs) ----
        if (active) {
            v16h A2, Dm;
            v16h wc = *reinterpret_cast<const v16h*>(&sW[FR_W2C0][lane][0]);
#pragma unroll
            for (int i = 0; i < 8; ++i) {
                const int k0 = (i >> 2) * 16 + halfk * 8 + (i & 3) * 2;
                h2 pr = *reinterpret_cast<const h2*>(&sH[wid][m * 34 + k0]);
#pragma unroll
                for (int j = 0; j < 2; ++j) {
                    const _Float16 v   = pr[j];
                    const bool     pos = !(v < (_Float16)0.0f);
                    A2[2 * i + j] = pos ? v : (_Float16)((float)v * 0.2f);
                    Dm[2 * i + j] = (pos ? (_Float16)1.0f : (_Float16)0.2f) * wc[2 * i + j];
                }
            }
            v16h bw2f   = *reinterpret_cast<const v16h*>(&sW[FR_W2   ][lane][0]);
            v16h bw1Tlo = *reinterpret_cast<const v16h*>(&sW[FR_W1TLO][lane][0]);
            v16h bw1Thi = *reinterpret_cast<const v16h*>(&sW[FR_W1THI][lane][0]);
            v8f co = {}, g0 = {}, g1 = {};
            co = __builtin_amdgcn_wmma_f32_16x16x32_f16(false, A2, false, bw2f,   (short)0, co, false, false);
            g0 = __builtin_amdgcn_wmma_f32_16x16x32_f16(false, Dm, false, bw1Tlo, (short)0, g0, false, false);
            g1 = __builtin_amdgcn_wmma_f32_16x16x32_f16(false, Dm, false, bw1Thi, (short)0, g1, false, false);
#pragma unroll
            for (int r = 0; r < 8; ++r) {
                const int  M  = r + halfk * 8;
                const long pp = base + M;
                if (pp < N) {
                    const float val = co[r] + b2n;
                    if (m == 0)      outSdf[pp] = val;
                    else if (m < 4)  outRgb[pp * 3 + (m - 1)] = 0.5f * (tanhf(val) + 1.0f);
                }
                sG[wid][M * 33 + m]      = g0[r];
                sG[wid][M * 33 + m + 16] = g1[r];
            }
        }
        __syncthreads();

        // ---------------- phase 4: nablas = 511 * <dH/dcoord, G>  -------------------
        if (active) {
            float g[16];
#pragma unroll
            for (int f = 0; f < 16; ++f) g[f] = sG[wid][m * 33 + halfk * 16 + f];
            float na[3] = {0.f, 0.f, 0.f};
#pragma unroll
            for (int pl = 0; pl < 3; ++pl) {
                const int a = axA[pl], b = axB[pl];
                const float fa = F[a], fb = F[b];
                const float ofb = 1.f - fb, ofa = 1.f - fa;
                const float* pb  = planes + pl * RES * RES * NFEAT + halfk * 16;
                const float* r00 = pb + (P0[a] * RES + P0[b]) * NFEAT;
                const float* r01 = pb + (P0[a] * RES + P1[b]) * NFEAT;
                const float* r10 = pb + (P1[a] * RES + P0[b]) * NFEAT;
                const float* r11 = pb + (P1[a] * RES + P1[b]) * NFEAT;
                float da = 0.f, db = 0.f;
#pragma unroll
                for (int c = 0; c < 4; ++c) {          // 4-feature chunks
                    float4 a00 = ld4(r00, c), a01 = ld4(r01, c);
                    float4 a10 = ld4(r10, c), a11 = ld4(r11, c);
                    da += ((a10.x - a00.x) * ofb + (a11.x - a01.x) * fb) * g[4*c+0];
                    db += ((a01.x - a00.x) * ofa + (a11.x - a10.x) * fa) * g[4*c+0];
                    da += ((a10.y - a00.y) * ofb + (a11.y - a01.y) * fb) * g[4*c+1];
                    db += ((a01.y - a00.y) * ofa + (a11.y - a10.y) * fa) * g[4*c+1];
                    da += ((a10.z - a00.z) * ofb + (a11.z - a01.z) * fb) * g[4*c+2];
                    db += ((a01.z - a00.z) * ofa + (a11.z - a10.z) * fa) * g[4*c+2];
                    da += ((a10.w - a00.w) * ofb + (a11.w - a01.w) * fb) * g[4*c+3];
                    db += ((a01.w - a00.w) * ofa + (a11.w - a10.w) * fa) * g[4*c+3];
                }
                na[a] += 511.0f * da;
                na[b] += 511.0f * db;
            }
#pragma unroll
            for (int d = 0; d < 3; ++d) {
                const float v = na[d] + __shfl_xor(na[d], 16, 32);
                if (halfk == 0 && base + m < N) outNab[(base + m) * 3 + d] = v;
            }
        }
        __syncthreads();
    }
}

extern "C" void kernel_launch(void* const* d_in, const int* in_sizes, int n_in,
                              void* d_out, int out_size, void* d_ws, size_t ws_size,
                              hipStream_t stream) {
    const float* x      = (const float*)d_in[0];
    const float* planes = (const float*)d_in[1];
    const float* W1     = (const float*)d_in[2];
    const float* b1     = (const float*)d_in[3];
    const float* W2     = (const float*)d_in[4];
    const float* b2     = (const float*)d_in[5];
    float* out = (float*)d_out;

    const long N = (long)in_sizes[0] / 3;
    const long pointsPerBlock = 16L * WAVES * TPW;
    const int  blocks = (int)((N + pointsPerBlock - 1) / pointsPerBlock);

    triplane_fused_kernel<<<blocks, 32 * WAVES, 0, stream>>>(x, planes, W1, b1, W2, b2, out, N);
}
